// GraphConvolution_35820027249377
// MI455X (gfx1250) — compile-verified, matrix operand order checked by reading, not run on hardware
//
#include <hip/hip_runtime.h>
#include <stdint.h>

#define NN 50000
#define EE 200000
#define F1 513
#define C1 256
#define F2 257
#define C2 64
#define CAP 1024   // LDS-staged src ids per node (max plausible degree << CAP)

typedef __attribute__((ext_vector_type(16))) __bf16 v16bf;
typedef __attribute__((ext_vector_type(8)))  float  v8f;

union TileBF { uint4 q[2]; v16bf v; };

// float -> bf16 round-to-nearest-even
__device__ __forceinline__ unsigned short f2bf(float f) {
  unsigned u = __float_as_uint(f);
  unsigned r = (u + 0x7fffu + ((u >> 16) & 1u)) >> 16;
  return (unsigned short)r;
}

// ---------------- CSR build (dst-grouped adjacency) ----------------

__global__ void k_zero_deg(int* deg) {
  int i = blockIdx.x * blockDim.x + threadIdx.x;
  if (i < NN) deg[i] = 0;
}

__global__ void k_hist(const long long* ei, int* deg) {
  int e = blockIdx.x * blockDim.x + threadIdx.x;
  if (e < EE) atomicAdd(&deg[(int)ei[EE + e]], 1);
}

// single-block exclusive scan over deg -> rowptr (and cursor copy)
__global__ void k_scan(const int* deg, int* rowptr, int* cur) {
  __shared__ int sm[256];
  __shared__ int carry;
  if (threadIdx.x == 0) carry = 0;
  __syncthreads();
  for (int base = 0; base < NN; base += 256) {
    int i = base + threadIdx.x;
    int v = (i < NN) ? deg[i] : 0;
    sm[threadIdx.x] = v;
    __syncthreads();
    for (int ofs = 1; ofs < 256; ofs <<= 1) {
      int t = (threadIdx.x >= (unsigned)ofs) ? sm[threadIdx.x - ofs] : 0;
      __syncthreads();
      sm[threadIdx.x] += t;
      __syncthreads();
    }
    if (i < NN) {
      int ex = carry + sm[threadIdx.x] - v;
      rowptr[i] = ex;
      cur[i] = ex;
    }
    __syncthreads();
    if (threadIdx.x == 255) carry += sm[255];
    __syncthreads();
  }
  if (threadIdx.x == 0) rowptr[NN] = carry;
}

__global__ void k_fill(const long long* ei, int* cur, int* srcl) {
  int e = blockIdx.x * blockDim.x + threadIdx.x;
  if (e < EE) {
    int dst = (int)ei[EE + e];
    int idx = atomicAdd(&cur[dst], 1);
    srcl[idx] = (int)ei[e];
  }
}

// ---------------- weight / input conversions ----------------

__global__ void k_prep_w(const float* W1rel, const float* W1root,
                         const float* W2rel, const float* W2root,
                         unsigned short* w1relT, unsigned short* w1rootT,
                         unsigned short* w2relT, unsigned short* w2rootT) {
  int i = blockIdx.x * blockDim.x + threadIdx.x;
  const int S1 = 512 * 256, S2 = 256 * 64;
  if (i < S1)                { int n = i / 512, k = i % 512; w1relT[i]  = f2bf(W1rel [k * 256 + n]); }
  else if (i < 2 * S1)       { int j = i - S1;      int n = j / 512, k = j % 512; w1rootT[j] = f2bf(W1root[k * 256 + n]); }
  else if (i < 2 * S1 + S2)  { int j = i - 2 * S1;  int n = j / 256, k = j % 256; w2relT[j]  = f2bf(W2rel [k * 64 + n]); }
  else if (i < 2*S1 + 2*S2)  { int j = i - 2*S1-S2; int n = j / 256, k = j % 256; w2rootT[j] = f2bf(W2root[k * 64 + n]); }
}

__global__ void k_cvt_x(const float* x, unsigned short* xbf) {
  size_t i = (size_t)blockIdx.x * blockDim.x + threadIdx.x;
  if (i < (size_t)NN * 512) xbf[i] = f2bf(x[i]);
}

// ---------------- layer-1 softmax aggregation: gather + online softmax ----------------
// one block per dst node; threads parallel over the 513 features; src ids staged in LDS.

__global__ void k_aggr1(const float* x, const int* rowptr, const int* srcl,
                        unsigned short* a1bf, float* agg1last) {
  __shared__ int ss[CAP];
  int n = blockIdx.x;
  int beg = rowptr[n];
  int deg = rowptr[n + 1] - beg;
  int dc = deg < CAP ? deg : CAP;
  for (int j = threadIdx.x; j < dc; j += blockDim.x) ss[j] = srcl[beg + j];
  __syncthreads();
  for (int f = threadIdx.x; f < F1; f += blockDim.x) {
    float m = -3.4e38f, d = 0.f, num = 0.f;
    for (int j = 0; j < deg; j++) {
      int s = (j < CAP) ? ss[j] : srcl[beg + j];
      float v = (f < 512) ? x[(size_t)s * 512 + f] : (float)s;
      if (v > m) {          // online softmax rescale
        float sc = __expf(m - v);
        d = d * sc + 1.f;
        num = num * sc + v;
        m = v;
      } else {
        float e = __expf(v - m);
        d += e;
        num += e * v;
      }
    }
    float a = (deg > 0) ? (num / fmaxf(d, 1e-16f)) : 0.f;
    if (f < 512) a1bf[(size_t)n * 512 + f] = f2bf(a);
    else         agg1last[n] = a;
  }
}

// ---------------- GEMM 1: h2 = lrelu(agg1@W1rel + b1 + h@W1root) ----------------
// block = 128 (4 waves); wave -> one 16x16 C tile; block tile 16 rows x 64 cols.

__global__ void k_gemm1(const unsigned short* a1bf, const unsigned short* xbf,
                        const float* agg1last,
                        const unsigned short* w1relT, const unsigned short* w1rootT,
                        const float* W1rel, const float* W1root, const float* b1,
                        float* h2) {
  int wave = threadIdx.x >> 5, lane = threadIdx.x & 31;
  int lm = lane & 15, hi = lane >> 4;
  int row0 = blockIdx.x * 16;
  int col0 = blockIdx.y * 64 + wave * 16;

  const unsigned short* arow = a1bf   + (size_t)(row0 + lm) * 512;
  const unsigned short* hrow = xbf    + (size_t)(row0 + lm) * 512;
  const unsigned short* brel = w1relT + (size_t)(col0 + lm) * 512;
  const unsigned short* brot = w1rootT+ (size_t)(col0 + lm) * 512;
  int ko = hi * 8;

  v8f acc = {0.f, 0.f, 0.f, 0.f, 0.f, 0.f, 0.f, 0.f};
  for (int k0 = 0; k0 < 512; k0 += 32) {
    TileBF ta, th, tr, tt;
    ta.q[0] = *(const uint4*)(arow + k0 + ko);
    ta.q[1] = *(const uint4*)(arow + k0 + 16 + ko);
    th.q[0] = *(const uint4*)(hrow + k0 + ko);
    th.q[1] = *(const uint4*)(hrow + k0 + 16 + ko);
    tr.q[0] = *(const uint4*)(brel + k0 + ko);
    tr.q[1] = *(const uint4*)(brel + k0 + 16 + ko);
    tt.q[0] = *(const uint4*)(brot + k0 + ko);
    tt.q[1] = *(const uint4*)(brot + k0 + 16 + ko);
    acc = __builtin_amdgcn_wmma_f32_16x16x32_bf16(false, ta.v, false, tr.v, (short)0, acc, false, false);
    acc = __builtin_amdgcn_wmma_f32_16x16x32_bf16(false, th.v, false, tt.v, (short)0, acc, false, false);
  }

  int n = col0 + lm;
  float wrel_l  = W1rel [512 * 256 + n];   // fp32 side-path for the node-index column
  float wroot_l = W1root[512 * 256 + n];
  float bias    = b1[n];
#pragma unroll
  for (int v = 0; v < 8; v++) {
    int m = row0 + hi * 8 + v;
    float val = acc[v] + bias + agg1last[m] * wrel_l + (float)m * wroot_l;
    val = val >= 0.f ? val : 0.01f * val;
    h2[(size_t)m * 256 + n] = val;
  }
}

// ---------------- layer-2 sum aggregation: gather ----------------

__global__ void k_cvt_h2(const float* h2, unsigned short* h2bf) {
  size_t i = (size_t)blockIdx.x * blockDim.x + threadIdx.x;
  if (i < (size_t)NN * 256) h2bf[i] = f2bf(h2[i]);
}

__global__ void k_aggr2(const float* h2, const int* rowptr, const int* srcl,
                        unsigned short* a2bf, float* agg2last) {
  __shared__ int ss[CAP];
  int n = blockIdx.x;
  int beg = rowptr[n];
  int deg = rowptr[n + 1] - beg;
  int dc = deg < CAP ? deg : CAP;
  for (int j = threadIdx.x; j < dc; j += blockDim.x) ss[j] = srcl[beg + j];
  __syncthreads();
  for (int f = threadIdx.x; f < F2; f += blockDim.x) {
    float acc = 0.f;
    for (int j = 0; j < deg; j++) {
      int s = (j < CAP) ? ss[j] : srcl[beg + j];
      acc += (f < 256) ? h2[(size_t)s * 256 + f] : (float)s;
    }
    if (f < 256) a2bf[(size_t)n * 256 + f] = f2bf(acc);
    else         agg2last[n] = acc;
  }
}

// ---------------- GEMM 2: h3 = lrelu(agg2@W2rel + b2 + h2@W2root) ----------------

__global__ void k_gemm2(const unsigned short* a2bf, const unsigned short* h2bf,
                        const float* agg2last,
                        const unsigned short* w2relT, const unsigned short* w2rootT,
                        const float* W2rel, const float* W2root, const float* b2,
                        float* h3) {
  int wave = threadIdx.x >> 5, lane = threadIdx.x & 31;
  int lm = lane & 15, hi = lane >> 4;
  int row0 = blockIdx.x * 16;
  int col0 = wave * 16;                 // 4 waves cover all 64 output cols

  const unsigned short* arow = a2bf   + (size_t)(row0 + lm) * 256;
  const unsigned short* hrow = h2bf   + (size_t)(row0 + lm) * 256;
  const unsigned short* brel = w2relT + (size_t)(col0 + lm) * 256;
  const unsigned short* brot = w2rootT+ (size_t)(col0 + lm) * 256;
  int ko = hi * 8;

  v8f acc = {0.f, 0.f, 0.f, 0.f, 0.f, 0.f, 0.f, 0.f};
  for (int k0 = 0; k0 < 256; k0 += 32) {
    TileBF ta, th, tr, tt;
    ta.q[0] = *(const uint4*)(arow + k0 + ko);
    ta.q[1] = *(const uint4*)(arow + k0 + 16 + ko);
    th.q[0] = *(const uint4*)(hrow + k0 + ko);
    th.q[1] = *(const uint4*)(hrow + k0 + 16 + ko);
    tr.q[0] = *(const uint4*)(brel + k0 + ko);
    tr.q[1] = *(const uint4*)(brel + k0 + 16 + ko);
    tt.q[0] = *(const uint4*)(brot + k0 + ko);
    tt.q[1] = *(const uint4*)(brot + k0 + 16 + ko);
    acc = __builtin_amdgcn_wmma_f32_16x16x32_bf16(false, ta.v, false, tr.v, (short)0, acc, false, false);
    acc = __builtin_amdgcn_wmma_f32_16x16x32_bf16(false, th.v, false, tt.v, (short)0, acc, false, false);
  }

  int n = col0 + lm;
  float wrel_l  = W2rel [256 * 64 + n];
  float wroot_l = W2root[256 * 64 + n];
  float bias    = b2[n];
#pragma unroll
  for (int v = 0; v < 8; v++) {
    int m = row0 + hi * 8 + v;
    float val = acc[v] + bias + agg2last[m] * wrel_l + (float)m * wroot_l;
    val = val >= 0.f ? val : 0.01f * val;
    h3[(size_t)m * 64 + n] = val;
  }
}

// ---------------- heads ----------------

__global__ void k_head(const float* h3, const float* Wp, const float* bp,
                       const float* Wr, const float* br, float* out) {
  int n = blockIdx.x * blockDim.x + threadIdx.x;
  if (n >= NN) return;
  const float* h = h3 + (size_t)n * 64;
  float pos[3], rot[4];
#pragma unroll
  for (int j = 0; j < 3; j++) pos[j] = bp[j] + (float)n * Wp[64 * 3 + j];
#pragma unroll
  for (int j = 0; j < 4; j++) rot[j] = br[j] + (float)n * Wr[64 * 4 + j];
  for (int k = 0; k < 64; k++) {
    float hv = h[k];
#pragma unroll
    for (int j = 0; j < 3; j++) pos[j] += hv * Wp[k * 3 + j];
#pragma unroll
    for (int j = 0; j < 4; j++) rot[j] += hv * Wr[k * 4 + j];
  }
  float s = rot[0]*rot[0] + rot[1]*rot[1] + rot[2]*rot[2] + rot[3]*rot[3];
  float inv = 1.f / fmaxf(sqrtf(s), 1e-12f);
  float* o = out + (size_t)n * 7;
  o[0] = pos[0]; o[1] = pos[1]; o[2] = pos[2];
  o[3] = rot[0] * inv; o[4] = rot[1] * inv; o[5] = rot[2] * inv; o[6] = rot[3] * inv;
}

// ---------------- launcher ----------------

static inline size_t align_up(size_t v, size_t a) { return (v + a - 1) & ~(a - 1); }

extern "C" void kernel_launch(void* const* d_in, const int* in_sizes, int n_in,
                              void* d_out, int out_size, void* d_ws, size_t ws_size,
                              hipStream_t stream) {
  const float* x      = (const float*)d_in[0];
  const float* W1rel  = (const float*)d_in[1];
  const float* b1     = (const float*)d_in[2];
  const float* W1root = (const float*)d_in[3];
  const float* W2rel  = (const float*)d_in[4];
  const float* b2     = (const float*)d_in[5];
  const float* W2root = (const float*)d_in[6];
  const float* Wp     = (const float*)d_in[7];
  const float* bp     = (const float*)d_in[8];
  const float* Wr     = (const float*)d_in[9];
  const float* br     = (const float*)d_in[10];
  const long long* ei = (const long long*)d_in[11];
  float* out = (float*)d_out;

  // workspace layout (byte offsets, 256B aligned)
  char* base = (char*)d_ws;
  size_t off = 0;
  int* deg      = (int*)(base + off); off = align_up(off + sizeof(int) * NN, 256);
  int* rowptr   = (int*)(base + off); off = align_up(off + sizeof(int) * (NN + 1), 256);
  int* cur      = (int*)(base + off); off = align_up(off + sizeof(int) * NN, 256);
  int* srcl     = (int*)(base + off); off = align_up(off + sizeof(int) * EE, 256);
  unsigned short* xbf   = (unsigned short*)(base + off); off = align_up(off + 2ull * NN * 512, 256);
  unsigned short* a1bf  = (unsigned short*)(base + off); off = align_up(off + 2ull * NN * 512, 256);
  float* agg1last = (float*)(base + off); off = align_up(off + sizeof(float) * NN, 256);
  float* h2       = (float*)(base + off); off = align_up(off + sizeof(float) * (size_t)NN * 256, 256);
  unsigned short* h2bf  = (unsigned short*)(base + off); off = align_up(off + 2ull * NN * 256, 256);
  unsigned short* a2bf  = (unsigned short*)(base + off); off = align_up(off + 2ull * NN * 256, 256);
  float* agg2last = (float*)(base + off); off = align_up(off + sizeof(float) * NN, 256);
  float* h3       = (float*)(base + off); off = align_up(off + sizeof(float) * (size_t)NN * 64, 256);
  unsigned short* w1relT  = (unsigned short*)(base + off); off = align_up(off + 2ull * 512 * 256, 256);
  unsigned short* w1rootT = (unsigned short*)(base + off); off = align_up(off + 2ull * 512 * 256, 256);
  unsigned short* w2relT  = (unsigned short*)(base + off); off = align_up(off + 2ull * 64 * 256, 256);
  unsigned short* w2rootT = (unsigned short*)(base + off); off = align_up(off + 2ull * 64 * 256, 256);

  const int B = 256;

  // CSR build
  k_zero_deg<<<(NN + B - 1) / B, B, 0, stream>>>(deg);
  k_hist<<<(EE + B - 1) / B, B, 0, stream>>>(ei, deg);
  k_scan<<<1, B, 0, stream>>>(deg, rowptr, cur);
  k_fill<<<(EE + B - 1) / B, B, 0, stream>>>(ei, cur, srcl);

  // conversions
  k_prep_w<<<(2 * 512 * 256 + 2 * 64 * 256) / B, B, 0, stream>>>(W1rel, W1root, W2rel, W2root,
                                                                 w1relT, w1rootT, w2relT, w2rootT);
  k_cvt_x<<<(unsigned)(((size_t)NN * 512 + B - 1) / B), B, 0, stream>>>(x, xbf);

  // layer 1
  k_aggr1<<<NN, B, 0, stream>>>(x, rowptr, srcl, a1bf, agg1last);
  k_gemm1<<<dim3(NN / 16, C1 / 64), 128, 0, stream>>>(a1bf, xbf, agg1last, w1relT, w1rootT,
                                                      W1rel, W1root, b1, h2);

  // layer 2
  k_cvt_h2<<<(unsigned)(((size_t)NN * 256 + B - 1) / B), B, 0, stream>>>(h2, h2bf);
  k_aggr2<<<NN, B, 0, stream>>>(h2, rowptr, srcl, a2bf, agg2last);
  k_gemm2<<<NN / 16, 128, 0, stream>>>(a2bf, h2bf, agg2last, w2relT, w2rootT,
                                       W2rel, W2root, b2, h3);

  // heads
  k_head<<<(NN + B - 1) / B, B, 0, stream>>>(h3, Wp, bp, Wr, br, out);
}
